// MambaEncoderLayer_48936857371344
// MI455X (gfx1250) — matrix operand order repaired and checked by reading, not compiled, not run on hardware
//
#include <hip/hip_runtime.h>
#include <hip/hip_bf16.h>
#include <stdint.h>

// ---------------- types ----------------
typedef __bf16 bf16_t;
typedef __bf16 v16bf __attribute__((ext_vector_type(16)));
typedef __bf16 v8bf  __attribute__((ext_vector_type(8)));
typedef float  v8f   __attribute__((ext_vector_type(8)));

// ---------------- problem constants ----------------
constexpr int BATCH = 2;
constexpr int CCH   = 256;
constexpr int H0 = 128, W0 = 128, H1 = 256, W1 = 64;
constexpr int AGG = 4;
constexpr int H0D = H0 / AGG, W0D = W0 / AGG;   // 32 x 32
constexpr int H1D = H1 / AGG, W1D = W1 / AGG;   // 64 x 16
constexpr int LSEQ = 4096;                      // 4 * 1024
constexpr int DI = 512;
constexpr int NST = 16;
constexpr int RRK = 16;
constexpr int KCONV = CCH * AGG * AGG;          // 4096
constexpr int MROWS = 4096;                     // conv-gemm rows (all spatial positions)

// ---------------- helpers ----------------
__device__ __forceinline__ bf16_t f2bf(float f) {
  union { float f; uint32_t u; } v; v.f = f;
  uint32_t r = v.u + 0x7FFFu + ((v.u >> 16) & 1u);
  uint16_t h = (uint16_t)(r >> 16);
  return __builtin_bit_cast(bf16_t, h);
}
__device__ __forceinline__ float silu_f(float x) { return x / (1.0f + __expf(-x)); }
__device__ __forceinline__ float softplus_f(float x) {
  return (x > 20.0f) ? x : log1pf(__expf(x));
}

// ---------------- generic bf16 WMMA GEMM:  C[M,N] = A[M,K] * W[N,K]^T (+bias) ----------------
// block = 256 threads = 8 waves; wave tile 32x32 (2x2 wmma frags); block tile 128x64.
// M must be a multiple of 128; K a multiple of 32; N guarded.
__global__ __launch_bounds__(256)
void gemm_bf16_wmma(const bf16_t* __restrict__ A, const bf16_t* __restrict__ W,
                    const float* __restrict__ bias, float* __restrict__ C,
                    int M, int N, int K)
{
  const int tid   = threadIdx.x;
  const int wave  = tid >> 5;
  const int lane  = tid & 31;
  const int lhalf = lane >> 4;      // 0/1 (K-half selector)
  const int l16   = lane & 15;
  const int wm    = wave & 3;       // 4 waves along M
  const int wn    = wave >> 2;      // 2 waves along N
  const int m0    = blockIdx.x * 128 + wm * 32;
  const int n0    = blockIdx.y * 64 + wn * 32;

  const v8f vzero = {0.f,0.f,0.f,0.f,0.f,0.f,0.f,0.f};
  v8f acc[2][2] = { {vzero, vzero}, {vzero, vzero} };

  // clamp out-of-range B columns to a valid row; those frag columns are never stored.
  int ncol[2];
  #pragma unroll
  for (int j = 0; j < 2; j++) {
    int n = n0 + 16 * j + l16;
    ncol[2 > j ? j : j] = (n < N) ? n : (N - 1);
  }
  const int arow[2] = { m0 + l16, m0 + 16 + l16 };

  for (int k0 = 0; k0 < K; k0 += 32) {
    if (k0 + 32 < K) {  // hint next tiles into cache (global_prefetch_b8)
      __builtin_prefetch((const void*)(A + (size_t)arow[0] * K + k0 + 32), 0, 3);
      __builtin_prefetch((const void*)(W + (size_t)ncol[0] * K + k0 + 32), 0, 3);
    }
    v16bf a[2], b[2];
    // A fragment (16-bit A 16x32 layout): lane half selects K sub-chunk of 8,
    // two separated 8-element runs -> two b128 loads per fragment.
    #pragma unroll
    for (int i = 0; i < 2; i++) {
      const bf16_t* p = A + (size_t)arow[i] * K + k0 + lhalf * 8;
      v8bf lo = *(const v8bf*)(p);
      v8bf hi = *(const v8bf*)(p + 16);
      #pragma unroll
      for (int e = 0; e < 8; e++) { a[i][e] = lo[e]; a[i][8 + e] = hi[e]; }
    }
    // B fragment (16-bit B 32x16 layout): lane half selects K sub-chunk of 8,
    // two separated 8-element runs (same K order as A).
    #pragma unroll
    for (int j = 0; j < 2; j++) {
      const bf16_t* p = W + (size_t)ncol[j] * K + k0 + lhalf * 8;
      v8bf lo = *(const v8bf*)(p);
      v8bf hi = *(const v8bf*)(p + 16);
      #pragma unroll
      for (int e = 0; e < 8; e++) { b[j][e] = lo[e]; b[j][8 + e] = hi[e]; }
    }
    #pragma unroll
    for (int i = 0; i < 2; i++)
      #pragma unroll
      for (int j = 0; j < 2; j++)
        acc[i][j] = __builtin_amdgcn_wmma_f32_16x16x32_bf16(
            false, a[i], false, b[j], (short)0, acc[i][j], false, false);
  }

  // store: 32-bit C/D 16x16 layout -> VGPR v: row = v + 8*lhalf; col = l16.
  #pragma unroll
  for (int j = 0; j < 2; j++) {
    int col = n0 + 16 * j + l16;
    if (col >= N) continue;
    float bv = bias ? bias[col] : 0.0f;
    #pragma unroll
    for (int i = 0; i < 2; i++) {
      #pragma unroll
      for (int v = 0; v < 8; v++) {
        int row = m0 + 16 * i + v + 8 * lhalf;
        C[(size_t)row * N + col] = acc[i][j][v] + bv;
      }
    }
  }
}

// ---------------- f32 -> bf16 convert ----------------
__global__ void cvt_f32_bf16(const float* __restrict__ src, bf16_t* __restrict__ dst, int n) {
  int i = blockIdx.x * blockDim.x + threadIdx.x;
  if (i < n) dst[i] = f2bf(src[i]);
}

// ---------------- im2col for 4x4/stride-4 patchify conv ----------------
// rows [0,2048): x0  (b,oh,ow) 32x32 ; rows [2048,4096): x1 (b,oh,ow) 64x16.
// k = ci*16 + kh*4 + kw  (matches conv_w [co][ci][kh][kw] flattening)
__global__ void im2col_kernel(const float* __restrict__ x0, const float* __restrict__ x1,
                              bf16_t* __restrict__ out)
{
  size_t idx = (size_t)blockIdx.x * blockDim.x + threadIdx.x;
  if (idx >= (size_t)MROWS * KCONV) return;
  int r = (int)(idx / KCONV);
  int k = (int)(idx % KCONV);
  int ci = k >> 4, kh = (k >> 2) & 3, kw = k & 3;
  float v;
  if (r < 2048) {
    int b = r >> 10, p = r & 1023;
    int oh = p >> 5, ow = p & 31;
    v = x0[(((size_t)b * CCH + ci) * H0 + (oh * 4 + kh)) * W0 + (ow * 4 + kw)];
  } else {
    int rr = r - 2048;
    int b = rr >> 10, p = rr & 1023;
    int oh = p >> 4, ow = p & 15;
    v = x1[(((size_t)b * CCH + ci) * H1 + (oh * 4 + kh)) * W1 + (ow * 4 + kw)];
  }
  out[idx] = f2bf(v);
}

// ---------------- layernorm over channels + build sequence layouts ----------------
// writes x_f (f32 + bf16) at both hw (base) and wh (2048+base) positions, and the
// batch-wise reversed bf16 copy for the backward direction.
__global__ __launch_bounds__(256)
void ln_build_kernel(const float* __restrict__ xdpre,
                     const float* __restrict__ lnw, const float* __restrict__ lnb,
                     float* __restrict__ xf, bf16_t* __restrict__ xf_bf,
                     bf16_t* __restrict__ xfrev_bf)
{
  int r = blockIdx.x;      // 0..4095
  int c = threadIdx.x;     // 0..255
  float x = xdpre[(size_t)r * CCH + c];
  __shared__ float red[256];
  red[c] = x; __syncthreads();
  for (int s = 128; s > 0; s >>= 1) { if (c < s) red[c] += red[c + s]; __syncthreads(); }
  float mean = red[0] * (1.0f / 256.0f);
  __syncthreads();
  float d = x - mean;
  red[c] = d * d; __syncthreads();
  for (int s = 128; s > 0; s >>= 1) { if (c < s) red[c] += red[c + s]; __syncthreads(); }
  float var = red[0] * (1.0f / 256.0f);
  float y = d * rsqrtf(var + 1e-6f) * lnw[c] + lnb[c];

  int b, base;
  if (r < 2048) { b = r >> 10; base = r & 1023; }
  else          { b = (r - 2048) >> 10; base = 1024 + ((r - 2048) & 1023); }
  size_t p1 = ((size_t)b * LSEQ + base) * CCH + c;
  size_t p2 = ((size_t)b * LSEQ + 2048 + base) * CCH + c;
  bf16_t yb = f2bf(y);
  xf[p1] = y;  xf[p2] = y;
  xf_bf[p1] = yb; xf_bf[p2] = yb;
  size_t q1 = ((size_t)b * LSEQ + (LSEQ - 1 - base)) * CCH + c;
  size_t q2 = ((size_t)b * LSEQ + (LSEQ - 1 - (2048 + base))) * CCH + c;
  xfrev_bf[q1] = yb; xfrev_bf[q2] = yb;
}

// ---------------- causal depthwise conv (K=4) + SiLU ----------------
__global__ void dwconv_kernel(const float* __restrict__ XZ,
                              const float* __restrict__ c1w, const float* __restrict__ c1b,
                              float* __restrict__ xc, bf16_t* __restrict__ xc_bf)
{
  int idx = blockIdx.x * blockDim.x + threadIdx.x;   // over B*L*DI = 2^22
  if (idx >= BATCH * LSEQ * DI) return;
  int d = idx & (DI - 1);
  int l = (idx >> 9) & (LSEQ - 1);
  int b = idx >> 21;
  float s = c1b[d];
  #pragma unroll
  for (int j = 0; j < 4; j++) {
    int ls = l - 3 + j;
    if (ls >= 0) s += XZ[((size_t)b * LSEQ + ls) * 1024 + d] * c1w[d * 4 + j];
  }
  float y = silu_f(s);
  xc[idx] = y;
  xc_bf[idx] = f2bf(y);
}

// ---------------- dt projection (K=16) + softplus ----------------
__global__ void dt_kernel(const float* __restrict__ xdb,
                          const float* __restrict__ dtw, const float* __restrict__ dtb,
                          float* __restrict__ dt)
{
  int idx = blockIdx.x * blockDim.x + threadIdx.x;   // over B*L*DI
  if (idx >= BATCH * LSEQ * DI) return;
  int d = idx & (DI - 1);
  int row = idx >> 9;                                 // b*L + l
  float s = dtb[d];
  #pragma unroll
  for (int j = 0; j < RRK; j++) s += xdb[(size_t)row * 48 + j] * dtw[d * 16 + j];
  dt[idx] = softplus_f(s);
}

// ---------------- selective scan: wave handles 2 channels, lane = state index n ----------------
__global__ __launch_bounds__(256)
void scan_kernel(const float* __restrict__ dt, const float* __restrict__ xc,
                 const float* __restrict__ xdb, const float* __restrict__ alog,
                 const float* __restrict__ dp, float* __restrict__ y)
{
  int wglobal = blockIdx.x * (blockDim.x >> 5) + (threadIdx.x >> 5);  // 0..511
  int lane = threadIdx.x & 31;
  int n = lane & 15;
  int grp = lane >> 4;
  int b = wglobal >> 8;           // 0..1
  int d = (wglobal & 255) * 2 + grp;
  float A  = -__expf(alog[d * 16 + n]);
  float Dv = dp[d];
  float h = 0.0f;
  for (int l = 0; l < LSEQ; l++) {
    size_t base = (size_t)b * LSEQ + l;
    float dtv = dt[base * DI + d];
    float u   = xc[base * DI + d];
    float Bv  = xdb[base * 48 + 16 + n];
    float Cv  = xdb[base * 48 + 32 + n];
    h = __expf(dtv * A) * h + (dtv * u) * Bv;
    float p = h * Cv;
    p += __shfl_xor(p, 1, 32);
    p += __shfl_xor(p, 2, 32);
    p += __shfl_xor(p, 4, 32);
    p += __shfl_xor(p, 8, 32);
    if (n == 0) y[base * DI + d] = p + u * Dv;
  }
}

// ---------------- gate: yz = y_scan * silu(z), to bf16 ----------------
__global__ void gate_kernel(const float* __restrict__ yscan, const float* __restrict__ XZ,
                            bf16_t* __restrict__ yz)
{
  int idx = blockIdx.x * blockDim.x + threadIdx.x;   // over B*L*DI
  if (idx >= BATCH * LSEQ * DI) return;
  int d = idx & (DI - 1);
  int row = idx >> 9;
  float z = XZ[(size_t)row * 1024 + DI + d];
  yz[idx] = f2bf(yscan[idx] * silu_f(z));
}

// ---------------- combine: ycomb = 0.5*(y_f + rev(y_b)) ----------------
__global__ void combine_kernel(const float* __restrict__ ydir, float* __restrict__ ycomb, int rev)
{
  int idx = blockIdx.x * blockDim.x + threadIdx.x;   // over B*L*C = 2^21
  if (idx >= BATCH * LSEQ * CCH) return;
  if (!rev) {
    ycomb[idx] = 0.5f * ydir[idx];
  } else {
    int c = idx & 255, l = (idx >> 8) & (LSEQ - 1), b = idx >> 20;
    ycomb[idx] += 0.5f * ydir[((size_t)b * LSEQ + (LSEQ - 1 - l)) * CCH + c];
  }
}

// ---------------- scatter + (hw+wh) sum + bilinear x4 upsample ----------------
__global__ void output_kernel(const float* __restrict__ ycomb, float* __restrict__ out)
{
  size_t idx = (size_t)blockIdx.x * blockDim.x + threadIdx.x;
  if (idx >= (size_t)16777216) return;
  int b, c, h, w, Hs, Ws, base_hw, base_wh;
  if (idx < (size_t)8388608) {                 // o0: (2,256,128,128)
    int t = (int)idx;
    w = t & 127; t >>= 7; h = t & 127; t >>= 7; c = t & 255; b = t >> 8;
    Hs = H0D; Ws = W0D; base_hw = 0; base_wh = 2048;
  } else {                                     // o1: (2,256,256,64)
    int t = (int)(idx - 8388608);
    w = t & 63; t >>= 6; h = t & 255; t >>= 8; c = t & 255; b = t >> 8;
    Hs = H1D; Ws = W1D; base_hw = 1024; base_wh = 3072;
  }
  auto V = [&](int hs, int ws) -> float {
    size_t phw = ((size_t)b * LSEQ + base_hw + hs * Ws + ws) * CCH + c;
    size_t pwh = ((size_t)b * LSEQ + base_wh + ws * Hs + hs) * CCH + c;
    return ycomb[phw] + ycomb[pwh];
  };
  float sy = (h + 0.5f) * 0.25f - 0.5f;
  float sx = (w + 0.5f) * 0.25f - 0.5f;
  int y0 = (int)floorf(sy); float fy = sy - (float)y0;
  int x0 = (int)floorf(sx); float fx = sx - (float)x0;
  int y0c = y0 < 0 ? 0 : (y0 > Hs - 1 ? Hs - 1 : y0);
  int y1c = (y0 + 1) < 0 ? 0 : ((y0 + 1) > Hs - 1 ? Hs - 1 : y0 + 1);
  int x0c = x0 < 0 ? 0 : (x0 > Ws - 1 ? Ws - 1 : x0);
  int x1c = (x0 + 1) < 0 ? 0 : ((x0 + 1) > Ws - 1 ? Ws - 1 : x0 + 1);
  float v = (1.f - fy) * ((1.f - fx) * V(y0c, x0c) + fx * V(y0c, x1c))
          +        fy  * ((1.f - fx) * V(y1c, x0c) + fx * V(y1c, x1c));
  out[idx] = v;
}

// ---------------- host launcher ----------------
extern "C" void kernel_launch(void* const* d_in, const int* in_sizes, int n_in,
                              void* d_out, int out_size, void* d_ws, size_t ws_size,
                              hipStream_t stream)
{
  (void)in_sizes; (void)n_in; (void)out_size; (void)ws_size;
  const float* x0     = (const float*)d_in[0];
  const float* x1     = (const float*)d_in[1];
  const float* conv_w = (const float*)d_in[2];
  const float* conv_b = (const float*)d_in[3];
  const float* ln_w   = (const float*)d_in[4];
  const float* ln_b   = (const float*)d_in[5];
  const float* in_w   = (const float*)d_in[6];
  const float* c1_w   = (const float*)d_in[7];
  const float* c1_b   = (const float*)d_in[8];
  const float* xp_w   = (const float*)d_in[9];
  const float* dt_w   = (const float*)d_in[10];
  const float* dt_b   = (const float*)d_in[11];
  const float* a_log  = (const float*)d_in[12];
  const float* d_p    = (const float*)d_in[13];
  const float* out_w  = (const float*)d_in[14];
  float* out = (float*)d_out;

  char* ws = (char*)d_ws;
  size_t off = 0;
  auto take = [&](size_t bytes) -> char* {
    char* p = ws + off;
    off += (bytes + 255) & ~(size_t)255;
    return p;
  };

  // union block: im2col(bf16 4096x4096) first, then reused as XZ(f32 8192x1024)
  char*   union0    = take(33554432);
  bf16_t* im2col    = (bf16_t*)union0;
  float*  XZ        = (float*)union0;
  bf16_t* convw_bf  = (bf16_t*)take((size_t)CCH * KCONV * 2);          // 2 MB
  bf16_t* inw_bf    = (bf16_t*)take((size_t)2 * 1024 * CCH * 2);       // 1 MB
  bf16_t* xpw_bf    = (bf16_t*)take((size_t)2 * 48 * DI * 2);
  bf16_t* outw_bf   = (bf16_t*)take((size_t)2 * CCH * DI * 2);
  float*  xdpre     = (float*)take((size_t)MROWS * CCH * 4);           // 4 MB
  float*  xf        = (float*)take((size_t)BATCH * LSEQ * CCH * 4);    // 8 MB
  bf16_t* xf_bf     = (bf16_t*)take((size_t)BATCH * LSEQ * CCH * 2);
  bf16_t* xfrev_bf  = (bf16_t*)take((size_t)BATCH * LSEQ * CCH * 2);
  float*  xc        = (float*)take((size_t)BATCH * LSEQ * DI * 4);     // 16 MB
  bf16_t* xc_bf     = (bf16_t*)take((size_t)BATCH * LSEQ * DI * 2);
  float*  xdb       = (float*)take((size_t)BATCH * LSEQ * 48 * 4);
  float*  dtbuf     = (float*)take((size_t)BATCH * LSEQ * DI * 4);
  float*  yscan     = (float*)take((size_t)BATCH * LSEQ * DI * 4);
  bf16_t* yz_bf     = (bf16_t*)take((size_t)BATCH * LSEQ * DI * 2);
  float*  ydir      = (float*)take((size_t)BATCH * LSEQ * CCH * 4);
  float*  ycomb     = (float*)take((size_t)BATCH * LSEQ * CCH * 4);

  const int TB = 256;
  auto blocks = [](size_t n, int tb) { return (unsigned)((n + tb - 1) / tb); };

  // 1) weight conversions to bf16
  cvt_f32_bf16<<<blocks((size_t)CCH * KCONV, TB), TB, 0, stream>>>(conv_w, convw_bf, CCH * KCONV);
  cvt_f32_bf16<<<blocks((size_t)2 * 1024 * CCH, TB), TB, 0, stream>>>(in_w, inw_bf, 2 * 1024 * CCH);
  cvt_f32_bf16<<<blocks((size_t)2 * 48 * DI, TB), TB, 0, stream>>>(xp_w, xpw_bf, 2 * 48 * DI);
  cvt_f32_bf16<<<blocks((size_t)2 * CCH * DI, TB), TB, 0, stream>>>(out_w, outw_bf, 2 * CCH * DI);

  // 2) patchify im2col + conv GEMM (+bias)
  im2col_kernel<<<blocks((size_t)MROWS * KCONV, TB), TB, 0, stream>>>(x0, x1, im2col);
  {
    dim3 g(MROWS / 128, (CCH + 63) / 64);
    gemm_bf16_wmma<<<g, TB, 0, stream>>>(im2col, convw_bf, conv_b, xdpre, MROWS, CCH, KCONV);
  }

  // 3) layernorm + sequence layout build
  ln_build_kernel<<<MROWS, TB, 0, stream>>>(xdpre, ln_w, ln_b, xf, xf_bf, xfrev_bf);

  // 4) two mamba directions (buffers reused; im2col region now reused as XZ)
  const int MSEQ = BATCH * LSEQ;            // 8192 rows
  for (int dir = 0; dir < 2; dir++) {
    const bf16_t* Adir = dir ? xfrev_bf : xf_bf;
    {   // in_proj: [8192,1024] = A[8192,256] * in_w[1024,256]^T
      dim3 g(MSEQ / 128, 1024 / 64);
      gemm_bf16_wmma<<<g, TB, 0, stream>>>(Adir, inw_bf + (size_t)dir * 1024 * CCH,
                                           nullptr, XZ, MSEQ, 1024, CCH);
    }
    dwconv_kernel<<<blocks((size_t)MSEQ * DI, TB), TB, 0, stream>>>(
        XZ, c1_w + (size_t)dir * DI * 4, c1_b + (size_t)dir * DI, xc, xc_bf);
    {   // x-proj: [8192,48] = xc[8192,512] * xp_w[48,512]^T
      dim3 g(MSEQ / 128, (48 + 63) / 64);
      gemm_bf16_wmma<<<g, TB, 0, stream>>>(xc_bf, xpw_bf + (size_t)dir * 48 * DI,
                                           nullptr, xdb, MSEQ, 48, DI);
    }
    dt_kernel<<<blocks((size_t)MSEQ * DI, TB), TB, 0, stream>>>(
        xdb, dt_w + (size_t)dir * DI * RRK, dt_b + (size_t)dir * DI, dtbuf);
    scan_kernel<<<64, TB, 0, stream>>>(dtbuf, xc, xdb,
        a_log + (size_t)dir * DI * NST, d_p + (size_t)dir * DI, yscan);
    gate_kernel<<<blocks((size_t)MSEQ * DI, TB), TB, 0, stream>>>(yscan, XZ, yz_bf);
    {   // out_proj: [8192,256] = yz[8192,512] * out_w[256,512]^T
      dim3 g(MSEQ / 128, CCH / 64);
      gemm_bf16_wmma<<<g, TB, 0, stream>>>(yz_bf, outw_bf + (size_t)dir * CCH * DI,
                                           nullptr, ydir, MSEQ, CCH, DI);
    }
    combine_kernel<<<blocks((size_t)MSEQ * CCH, TB), TB, 0, stream>>>(ydir, ycomb, dir);
  }

  // 5) scatter + sum(hw,wh) + bilinear x4 upsample into d_out (o0 then o1)
  output_kernel<<<blocks((size_t)16777216, TB), TB, 0, stream>>>(ycomb, out);
}
